// TransformerGSM_57655640981773
// MI455X (gfx1250) — matrix operand order repaired and checked by reading, not compiled, not run on hardware
//
#include <hip/hip_runtime.h>
#include <hip/hip_bf16.h>
#include <math.h>

typedef __attribute__((ext_vector_type(4)))  float     v4f;
typedef __attribute__((ext_vector_type(8)))  float     v8f;
typedef __attribute__((ext_vector_type(4)))  _Float16  v4h;
typedef __attribute__((ext_vector_type(8)))  _Float16  v8h;
typedef __attribute__((ext_vector_type(16))) _Float16  v16h;

// ---------------------------------------------------------------------------
// Kernel 1: streaming copy of the whole tensor (patch tokens dominate; the 512
// cls rows are overwritten by kernel 2). Non-temporal b128 loads/stores:
// 620 MB of traffic >> 192 MB L2, so bypassing cache retention is correct.
// Roofline: ~620 MB @ 23.3 TB/s ~= 27 us; this kernel is the whole runtime.
// ---------------------------------------------------------------------------
__global__ __launch_bounds__(256) void gsm_bulk_copy(const v4f* __restrict__ src,
                                                     v4f* __restrict__ dst,
                                                     long n4) {
    long i = (long)blockIdx.x * blockDim.x + threadIdx.x;
    const long stride = (long)gridDim.x * blockDim.x;
    for (; i < n4; i += stride) {
        v4f d = __builtin_nontemporal_load(&src[i]);
        __builtin_nontemporal_store(d, &dst[i]);
    }
}

// ---------------------------------------------------------------------------
// Kernel 2: cls-token path. One block per batch b; 16 waves = 16 time steps.
//   wave t: LayerNorm+ReLU of cls row (b,t) -> h (f16) into LDS
//   WMMA:   (16 tokens x 768) @ (768 x 16; cols 0,1 = fc_w, rest zero-padded)
//   combine: out[c] = x_t[c]*(1-gate) + gate_neighbor * x_{t+-1}[c]
// B is staged zero-padded in LDS so the WMMA inner loop is pure
// ds_load_b128 -> v_wmma with no VALU masking in between.
// ---------------------------------------------------------------------------
#define T_FRAMES 16
#define N_TOK    197
#define C_DIM    768
#define C_HALF   384

__global__ __launch_bounds__(512) void gsm_cls_kernel(
    const float* __restrict__ x,
    const float* __restrict__ gamma,
    const float* __restrict__ beta,
    const float* __restrict__ fcw,   // (2, 768)
    const float* __restrict__ fcb,   // (2,)
    float* __restrict__ out) {

    __shared__ __attribute__((aligned(16))) _Float16 h_lds[T_FRAMES][C_DIM];
    __shared__ __attribute__((aligned(16))) _Float16 w_lds[16][C_DIM]; // rows 2..15 = 0
    __shared__ float g1g[T_FRAMES];
    __shared__ float g2g[T_FRAMES];

    const int b    = blockIdx.x;
    const int t    = threadIdx.x >> 5;   // wave id = time step
    const int lane = threadIdx.x & 31;

    const float* xrow = x + ((size_t)(b * T_FRAMES + t) * N_TOK) * C_DIM; // cls row

    // Stage fc_w into LDS rows 0..1 as f16; zero rows 2..15 (B zero-padding).
    for (int i = threadIdx.x; i < 2 * C_DIM; i += blockDim.x)
        w_lds[0][i] = (_Float16)fcw[i];
    for (int i = threadIdx.x; i < 14 * C_DIM / 4; i += blockDim.x)
        *(v4h*)&w_lds[2][i * 4] = (v4h){};

    // ---- LayerNorm + ReLU (one wave per token; 24 channels per lane) ----
    float v[24];
    float s = 0.f, sq = 0.f;
#pragma unroll
    for (int ch = 0; ch < 6; ++ch) {
        v4f d = *(const v4f*)(xrow + ch * 128 + lane * 4);
#pragma unroll
        for (int j = 0; j < 4; ++j) {
            float f = d[j];
            v[ch * 4 + j] = f;
            s += f;
            sq += f * f;
        }
    }
#pragma unroll
    for (int off = 16; off > 0; off >>= 1) {
        s  += __shfl_xor(s,  off, 32);
        sq += __shfl_xor(sq, off, 32);
    }
    const float mu = s * (1.0f / 768.0f);
    const float rs = rsqrtf(sq * (1.0f / 768.0f) - mu * mu + 1e-5f);

#pragma unroll
    for (int ch = 0; ch < 6; ++ch) {
        v4f gm = *(const v4f*)(gamma + ch * 128 + lane * 4);
        v4f bt = *(const v4f*)(beta  + ch * 128 + lane * 4);
        v4h hh;
#pragma unroll
        for (int j = 0; j < 4; ++j) {
            float n = (v[ch * 4 + j] - mu) * rs * gm[j] + bt[j];
            hh[j] = (_Float16)fmaxf(n, 0.0f);
        }
        *(v4h*)&h_lds[t][ch * 128 + lane * 4] = hh;
    }
    __syncthreads();

    // ---- WMMA gate GEMM: D(16x16) += A(16x32 f16) x B(32x16 f16) over 24 k-steps
    // A layout (16-bit 16x32): lane<16 -> M=lane, halves = K 0..7 | 16..23
    //                          lane>=16 -> M=lane-16, halves = K 8..15 | 24..31
    // B layout (32x16):        lane<16 -> N=lane, halves = K 0..7 | 16..23
    //                          lane>=16 -> N=lane-16, halves = K 8..15 | 24..31
    const int tok   = lane & 15;
    const int aoff  = (lane < 16) ? 0 : 8;
    const int koffB = (lane < 16) ? 0 : 8;
    const _Float16* wrow = &w_lds[lane & 15][0];   // rows >= 2 are zero

    v8f acc = {};
#pragma unroll
    for (int kk = 0; kk < 24; ++kk) {
        const int kbase = kk * 32;
        v8h alo = *(const v8h*)&h_lds[tok][kbase + aoff];
        v8h ahi = *(const v8h*)&h_lds[tok][kbase + aoff + 16];
        v8h blo = *(const v8h*)&wrow[kbase + koffB];
        v8h bhi = *(const v8h*)&wrow[kbase + koffB + 16];
        v16h a, bm;
#pragma unroll
        for (int j = 0; j < 8; ++j) {
            a[j]      = alo[j];
            a[8 + j]  = ahi[j];
            bm[j]     = blo[j];
            bm[8 + j] = bhi[j];
        }
        acc = __builtin_amdgcn_wmma_f32_16x16x32_f16(
            /*neg_a=*/false, a, /*neg_b=*/false, bm,
            /*c_mod=*/(short)0, acc, /*reuse_a=*/false, /*reuse_b=*/false);
    }

    // D layout: lanes 0..15 -> N=lane, VGPR r = row M=r; lanes 16..31 -> M=8+r.
    // Column 0 = gate1 pre-act, column 1 = gate2 pre-act. Wave 0 publishes.
    if (threadIdx.x < 32) {
        const float bias = fcb[lane & 1];
        const int mbase = (lane & 16) ? 8 : 0;
        if ((lane & 14) == 0) {            // lanes 0, 1, 16, 17
            float* dst = (lane & 1) ? g2g : g1g;
#pragma unroll
            for (int r = 0; r < 8; ++r)
                dst[mbase + r] = tanhf(acc[r] + bias);
        }
    }
    __syncthreads();

    // ---- Gated temporal shift-mix; identity channel mapping.
    // c <  384: out = x_t[c]*(1-g2[t]) + g2[t-1]*x_{t-1}[c]
    // c >= 384: out = x_t[c]*(1-g1[t]) + g1[t+1]*x_{t+1}[c]
    const float g2t = g2g[t];
    const float g1t = g1g[t];
    const float g2p = (t > 0)             ? g2g[t - 1] : 0.0f;
    const float g1n = (t < T_FRAMES - 1)  ? g1g[t + 1] : 0.0f;

    const float* xprev = xrow - (size_t)N_TOK * C_DIM;
    const float* xnext = xrow + (size_t)N_TOK * C_DIM;
    float* orow = out + ((size_t)(b * T_FRAMES + t) * N_TOK) * C_DIM;

    v4f zero = {};
#pragma unroll
    for (int ch = 0; ch < 3; ++ch) {       // chunks 0..2 -> c < 384
        const int c0 = ch * 128 + lane * 4;
        v4f cur  = *(const v4f*)(xrow + c0);
        v4f prev = (t > 0) ? *(const v4f*)(xprev + c0) : zero;
        v4f res;
#pragma unroll
        for (int j = 0; j < 4; ++j)
            res[j] = cur[j] * (1.0f - g2t) + g2p * prev[j];
        *(v4f*)(orow + c0) = res;
    }
#pragma unroll
    for (int ch = 3; ch < 6; ++ch) {       // chunks 3..5 -> c >= 384
        const int c0 = ch * 128 + lane * 4;
        v4f cur = *(const v4f*)(xrow + c0);
        v4f nxt = (t < T_FRAMES - 1) ? *(const v4f*)(xnext + c0) : zero;
        v4f res;
#pragma unroll
        for (int j = 0; j < 4; ++j)
            res[j] = cur[j] * (1.0f - g1t) + g1n * nxt[j];
        *(v4f*)(orow + c0) = res;
    }
}

extern "C" void kernel_launch(void* const* d_in, const int* in_sizes, int n_in,
                              void* d_out, int out_size, void* d_ws, size_t ws_size,
                              hipStream_t stream) {
    const float* x     = (const float*)d_in[0];
    const float* gamma = (const float*)d_in[1];
    const float* beta  = (const float*)d_in[2];
    const float* fcw   = (const float*)d_in[3];
    const float* fcb   = (const float*)d_in[4];
    float* out = (float*)d_out;

    const long n4 = (long)512 * N_TOK * C_DIM / 4;   // 19,365,888 float4s
    gsm_bulk_copy<<<4096, 256, 0, stream>>>((const v4f*)x, (v4f*)out, n4);

    const int B = 512 / T_FRAMES;                    // 32 blocks, 16 waves each
    gsm_cls_kernel<<<B, 512, 0, stream>>>(x, gamma, beta, fcw, fcb, out);
}
